// SampleNet_59012850647118
// MI455X (gfx1250) — matrix operand order repaired and checked by reading, not compiled, not run on hardware
//
#include <hip/hip_runtime.h>

// ---------------------------------------------------------------------------
// CDNA5 (gfx1250) WMMA-based implementation of the kernel-predicting denoiser.
// Activations: NHWC bf16 (channel dim padded to multiple of 32).
// Convs (1x1 and 3x3) run as implicit GEMM on v_wmma_f32_16x16x32_bf16 with
// a 2x2 register tile per wave (32 out-ch x 32 pixels): 4 wmma per K-step on
// 2 A-fragments + 2 B-fragments (A reused across pixel tiles, B across ocTiles).
// ---------------------------------------------------------------------------

typedef __attribute__((ext_vector_type(16))) __bf16 v16bf;
typedef __attribute__((ext_vector_type(8)))  float  v8f;

union Frag16 {
    v16bf v;
    unsigned short u[16];
    uint4 q[2];
};

__device__ __forceinline__ unsigned short f2bf(float f) {
    unsigned int u = __builtin_bit_cast(unsigned int, f);
    unsigned int r = u + 0x7FFFu + ((u >> 16) & 1u);   // round-to-nearest-even
    return (unsigned short)(r >> 16);
}
__device__ __forceinline__ float bf2f(unsigned short h) {
    unsigned int u = ((unsigned int)h) << 16;
    return __builtin_bit_cast(float, u);
}

// ---------------------------------------------------------------------------
// Weight packing: OIHW fp32 -> WMMA A-fragment order, bf16.
// Layout: [tap][ocTile][kBlock][lane 0..31][elem 0..15], 32B contiguous/lane.
// A-matrix 16x32 bf16 per ISA: lane M = lane%16;
//   K = (v>=4?16:0) + (lane>=16?8:0) + (v%4)*2 + h   for elem e = 2v+h.
// ---------------------------------------------------------------------------
__global__ void pack_w_kernel(const float* __restrict__ w,
                              unsigned short* __restrict__ wpk,
                              int Cout, int Cin, int KB, int ocTiles, int taps)
{
    long long idx = (long long)blockIdx.x * blockDim.x + threadIdx.x;
    long long total = (long long)taps * ocTiles * KB * 512;
    if (idx >= total) return;
    int e    = (int)(idx & 15);
    int lane = (int)((idx >> 4) & 31);
    long long r = idx >> 9;
    int kb  = (int)(r % KB);  r /= KB;
    int ocT = (int)(r % ocTiles);
    int t   = (int)(r / ocTiles);
    int oc = (ocT << 4) + (lane & 15);
    int v = e >> 1, h = e & 1;
    int K = ((v >= 4) ? 16 : 0) + ((lane >= 16) ? 8 : 0) + ((v & 3) << 1) + h;
    int ci = (kb << 5) + K;
    float val = 0.f;
    if (oc < Cout && ci < Cin)
        val = w[((size_t)oc * Cin + ci) * taps + t];
    wpk[idx] = f2bf(val);
}

// ---------------------------------------------------------------------------
// Generic conv: taps==1 -> 1x1 GEMM, taps==9 -> 3x3 SAME implicit GEMM.
// Grid: x = pixel-pair tiles / 8 waves, y = batch, z = CoutPad/32.
// ---------------------------------------------------------------------------
__global__ __launch_bounds__(256)
void conv_wmma_kernel(const unsigned short* __restrict__ in,   // [B,H,W,KB*32]
                      const unsigned short* __restrict__ wpk,  // packed weights
                      const float* __restrict__ bias,
                      int CoutTrue,
                      unsigned short* __restrict__ out,        // [B,H,W,CoutPad]
                      int H, int W, int logW, int KB, int ocTiles, int CoutPad,
                      int taps, int relu)
{
    int wave = threadIdx.x >> 5;
    int lane = threadIdx.x & 31;
    int nPxPairs = (H * W) >> 5;
    int pxPair = blockIdx.x * (blockDim.x >> 5) + wave;
    if (pxPair >= nPxPairs) return;

    int b = blockIdx.y;
    int ocT0 = blockIdx.z << 1;                 // two adjacent 16-row oc tiles
    int col = lane & 15, kgrp = lane >> 4;
    int p0 = (pxPair << 5) + col;
    int p1 = p0 + 16;
    int x0 = p0 & (W - 1), y0 = p0 >> logW;
    int x1 = p1 & (W - 1), y1 = p1 >> logW;
    int HW = H * W;
    int CinPad = KB << 5;

    v8f acc00, acc01, acc10, acc11;             // [ocTile][pxTile]
#pragma unroll
    for (int v = 0; v < 8; ++v) {
        int oc0 = (ocT0 << 4) + v + (kgrp << 3);
        int oc1 = oc0 + 16;
        float b0 = (oc0 < CoutTrue) ? bias[oc0] : 0.f;
        float b1 = (oc1 < CoutTrue) ? bias[oc1] : 0.f;
        acc00[v] = b0; acc01[v] = b0;
        acc10[v] = b1; acc11[v] = b1;
    }

    const unsigned short* inB = in + (size_t)b * HW * CinPad + (kgrp << 3);
    const int wStride = KB << 9;                // 512 * KB elems per ocTile

    for (int t = 0; t < taps; ++t) {
        int dy = (taps == 9) ? (t / 3 - 1) : 0;
        int dx = (taps == 9) ? (t % 3 - 1) : 0;
        int xx0 = x0 + dx, yy0 = y0 + dy;
        int xx1 = x1 + dx, yy1 = y1 + dy;
        bool valid0 = (xx0 >= 0) && (xx0 < W) && (yy0 >= 0) && (yy0 < H);
        bool valid1 = (xx1 >= 0) && (xx1 < W) && (yy1 >= 0) && (yy1 < H);
        const unsigned short* src0 = inB + (size_t)(yy0 * W + xx0) * CinPad;
        const unsigned short* src1 = inB + (size_t)(yy1 * W + xx1) * CinPad;
        const unsigned short* wrow0 =
            wpk + (size_t)(t * ocTiles + ocT0) * wStride + lane * 16;
        const unsigned short* wrow1 = wrow0 + wStride;

        for (int kb = 0; kb < KB; ++kb) {
            Frag16 bf0, bf1, af0, af1;
            if (valid0) {
                bf0.q[0] = *(const uint4*)(src0);
                bf0.q[1] = *(const uint4*)(src0 + 16);
            } else {
                bf0.q[0] = uint4{0u, 0u, 0u, 0u};
                bf0.q[1] = uint4{0u, 0u, 0u, 0u};
            }
            if (valid1) {
                bf1.q[0] = *(const uint4*)(src1);
                bf1.q[1] = *(const uint4*)(src1 + 16);
            } else {
                bf1.q[0] = uint4{0u, 0u, 0u, 0u};
                bf1.q[1] = uint4{0u, 0u, 0u, 0u};
            }
            af0.q[0] = *(const uint4*)(wrow0);
            af0.q[1] = *(const uint4*)(wrow0 + 8);
            af1.q[0] = *(const uint4*)(wrow1);
            af1.q[1] = *(const uint4*)(wrow1 + 8);
            if (kb + 1 < KB) __builtin_prefetch(wrow0 + 512, 0, 1);
            acc00 = __builtin_amdgcn_wmma_f32_16x16x32_bf16(
                false, af0.v, false, bf0.v, (short)0, acc00, false, false);
            acc01 = __builtin_amdgcn_wmma_f32_16x16x32_bf16(
                false, af0.v, false, bf1.v, (short)0, acc01, false, false);
            acc10 = __builtin_amdgcn_wmma_f32_16x16x32_bf16(
                false, af1.v, false, bf0.v, (short)0, acc10, false, false);
            acc11 = __builtin_amdgcn_wmma_f32_16x16x32_bf16(
                false, af1.v, false, bf1.v, (short)0, acc11, false, false);
            src0 += 32; src1 += 32;
            wrow0 += 512; wrow1 += 512;
        }
    }

    unsigned short* d0 = out + (size_t)(b * HW + p0) * CoutPad;
    unsigned short* d1 = out + (size_t)(b * HW + p1) * CoutPad;
#pragma unroll
    for (int v = 0; v < 8; ++v) {
        int oc0 = (ocT0 << 4) + v + (kgrp << 3);
        int oc1 = oc0 + 16;
        float f00 = acc00[v], f01 = acc01[v], f10 = acc10[v], f11 = acc11[v];
        if (relu) {
            f00 = f00 > 0.f ? f00 : 0.f;
            f01 = f01 > 0.f ? f01 : 0.f;
            f10 = f10 > 0.f ? f10 : 0.f;
            f11 = f11 > 0.f ? f11 : 0.f;
        }
        d0[oc0] = f2bf(f00);
        d1[oc0] = f2bf(f01);
        d0[oc1] = f2bf(f10);
        d1[oc1] = f2bf(f11);
    }
}

// ---------------------------------------------------------------------------
// Temporal stats: mean/var over S of 18 guide channels (rgb tonemapped),
// writes bf16 NHWC stride-64 (channels 36..63 zero), plus fp32 radiance mean.
// ---------------------------------------------------------------------------
__global__ __launch_bounds__(256)
void stats_kernel(const float* __restrict__ color, const float* __restrict__ nd,
                  const float* __restrict__ alb,   const float* __restrict__ spec,
                  const float* __restrict__ uvt,   const float* __restrict__ mv,
                  unsigned short* __restrict__ out, float* __restrict__ radmean,
                  int S, int HW)
{
    int p = blockIdx.x * blockDim.x + threadIdx.x;
    if (p >= HW) return;
    int b = blockIdx.y;
    float sum[18], sq[18], rm[3] = {0.f, 0.f, 0.f};
#pragma unroll
    for (int i = 0; i < 18; ++i) { sum[i] = 0.f; sq[i] = 0.f; }
    for (int s = 0; s < S; ++s) {
        float v[18];
#pragma unroll
        for (int c = 0; c < 3; ++c) {
            float cv = color[((size_t)((b * S + s) * 3 + c)) * HW + p];
            rm[c] += cv;
            v[c] = log1pf(fmaxf(cv, 0.f));
        }
#pragma unroll
        for (int c = 0; c < 4; ++c) v[3 + c]  = nd  [((size_t)((b * S + s) * 4 + c)) * HW + p];
#pragma unroll
        for (int c = 0; c < 3; ++c) v[7 + c]  = alb [((size_t)((b * S + s) * 3 + c)) * HW + p];
#pragma unroll
        for (int c = 0; c < 3; ++c) v[10 + c] = spec[((size_t)((b * S + s) * 3 + c)) * HW + p];
#pragma unroll
        for (int c = 0; c < 3; ++c) v[13 + c] = uvt [((size_t)((b * S + s) * 3 + c)) * HW + p];
#pragma unroll
        for (int c = 0; c < 2; ++c) v[16 + c] = mv  [((size_t)((b * S + s) * 2 + c)) * HW + p];
#pragma unroll
        for (int i = 0; i < 18; ++i) { sum[i] += v[i]; sq[i] += v[i] * v[i]; }
    }
    unsigned short* o = out + (size_t)(b * HW + p) * 64;
    float inv = 1.f / (float)S;
#pragma unroll
    for (int i = 0; i < 18; ++i) {
        float m = sum[i] * inv;
        float var = sq[i] * inv - m * m;
        o[i] = f2bf(m);
        o[18 + i] = f2bf(var);
    }
#pragma unroll
    for (int i = 36; i < 64; ++i) o[i] = 0;
    float* r = radmean + (size_t)(b * HW + p) * 3;
#pragma unroll
    for (int c = 0; c < 3; ++c) r[c] = rm[c] * inv;
}

// 2x2 max pool, NHWC bf16.
__global__ __launch_bounds__(256)
void pool_kernel(const unsigned short* __restrict__ in,
                 unsigned short* __restrict__ out, int Ho, int Wo, int C)
{
    int i = blockIdx.x * blockDim.x + threadIdx.x;
    int tot = Ho * Wo * C;
    if (i >= tot) return;
    int b = blockIdx.y;
    int c = i % C;
    int r = i / C;
    int xo = r % Wo;
    int yo = r / Wo;
    int Hi = Ho * 2, Wi = Wo * 2;
    const unsigned short* ib = in + (size_t)b * Hi * Wi * C;
    float m = -1e30f;
#pragma unroll
    for (int dy = 0; dy < 2; ++dy)
#pragma unroll
        for (int dx = 0; dx < 2; ++dx)
            m = fmaxf(m, bf2f(ib[((size_t)(2 * yo + dy) * Wi + (2 * xo + dx)) * C + c]));
    out[((size_t)b * Ho * Wo + yo * Wo + xo) * C + c] = f2bf(m);
}

// Concat [maybe-2x-nearest-upsampled src (Cu ch), skip (Cs ch)] -> out (Cu+Cs).
__global__ __launch_bounds__(256)
void upcat_kernel(const unsigned short* __restrict__ up, int Cu,
                  const unsigned short* __restrict__ sk, int Cs,
                  unsigned short* __restrict__ out, int Hh, int Wh, int upshift)
{
    int C = Cu + Cs;
    int i = blockIdx.x * blockDim.x + threadIdx.x;
    int tot = Hh * Wh * C;
    if (i >= tot) return;
    int b = blockIdx.y;
    int c = i % C;
    int r = i / C;
    int x = r % Wh;
    int y = r / Wh;
    unsigned short val;
    if (c < Cu) {
        int Hu = Hh >> upshift, Wu = Wh >> upshift;
        val = up[((size_t)b * Hu * Wu + (y >> upshift) * Wu + (x >> upshift)) * Cu + c];
    } else {
        val = sk[((size_t)b * Hh * Wh + y * Wh + x) * Cs + (c - Cu)];
    }
    out[((size_t)b * Hh * Wh + y * Wh + x) * C + c] = val;
}

// exp(pw - max) over 289 channels per pixel; bf16 (stride Kpad) -> fp32 (stride K).
__global__ __launch_bounds__(256)
void softmax_kernel(const unsigned short* __restrict__ pw, float* __restrict__ wb,
                    int HW, int K, int Kpad)
{
    int p = blockIdx.x * blockDim.x + threadIdx.x;
    if (p >= HW) return;
    int b = blockIdx.y;
    const unsigned short* src = pw + (size_t)(b * HW + p) * Kpad;
    float mx = -1e30f;
    for (int k = 0; k < K; ++k) mx = fmaxf(mx, bf2f(src[k]));
    float* d = wb + (size_t)(b * HW + p) * K;
    for (int k = 0; k < K; ++k) d[k] = __expf(bf2f(src[k]) - mx);
}

// 17x17 per-pixel filter of radiance mean + normalization. Output NCHW fp32.
__global__ __launch_bounds__(256)
void wfilter_kernel(const float* __restrict__ rad, const float* __restrict__ wb,
                    float* __restrict__ out, int H, int W)
{
    int p = blockIdx.x * blockDim.x + threadIdx.x;
    int HW = H * W;
    if (p >= HW) return;
    int b = blockIdx.y;
    int x = p % W, y = p / W;
    const float* wp = wb + (size_t)(b * HW + p) * 289;
    float c0 = 0.f, c1 = 0.f, c2 = 0.f, ws = 0.f;
    for (int ky = 0; ky < 17; ++ky) {
        int yy = y + ky - 8;
        if (yy < 0 || yy >= H) continue;
        for (int kx = 0; kx < 17; ++kx) {
            int xx = x + kx - 8;
            if (xx < 0 || xx >= W) continue;
            float g = wp[ky * 17 + kx];
            const float* r = rad + ((size_t)b * HW + (size_t)yy * W + xx) * 3;
            c0 += g * r[0]; c1 += g * r[1]; c2 += g * r[2];
            ws += g;
        }
    }
    float inv = 1.f / (ws + 1e-6f);
    out[((size_t)(b * 3 + 0)) * HW + p] = c0 * inv;
    out[((size_t)(b * 3 + 1)) * HW + p] = c1 * inv;
    out[((size_t)(b * 3 + 2)) * HW + p] = c2 * inv;
}

// ---------------------------------------------------------------------------
// Host orchestration
// ---------------------------------------------------------------------------
struct Layer {
    unsigned short* wpk;
    const float* bias;
    int Cin, Cout, taps, KB, ocT, CoutPad;
};

extern "C" void kernel_launch(void* const* d_in, const int* in_sizes, int n_in,
                              void* d_out, int out_size, void* d_ws, size_t ws_size,
                              hipStream_t stream)
{
    (void)n_in; (void)out_size; (void)ws_size;
    const int B = 2, S = 8, H = 256, W = 256;
    const int HW = H * W;

    const float* color = (const float*)d_in[0];
    const float* nd    = (const float*)d_in[1];
    const float* alb   = (const float*)d_in[2];
    const float* spec  = (const float*)d_in[3];
    const float* uvt   = (const float*)d_in[4];
    const float* mv    = (const float*)d_in[5];

    // Param leaf ordering: JAX pytrees sort dict keys ('bott','dec','enc','kg','pr');
    // fall back to insertion order ('pr','kg','enc','bott','dec') if sizes disagree.
    int bp, bk, be, bb, bd;
    if (in_sizes[6] == 512 * 512 * 9) { bb = 6; bd = 8; be = 28; bk = 40; bp = 46; }
    else                              { bp = 6; bk = 12; be = 18; bb = 30; bd = 32; }

    char* base = (char*)d_ws;
    size_t off = 0;
    auto alloc = [&](size_t bytes) -> void* {
        void* r = base + off;
        off = (off + bytes + 255) & ~(size_t)255;
        return r;
    };

    auto mk = [&](int wi, int bi, int Cout, int Cin, int taps) -> Layer {
        Layer L;
        L.Cin = Cin; L.Cout = Cout; L.taps = taps;
        L.KB = (Cin + 31) / 32;
        L.CoutPad = ((Cout + 31) / 32) * 32;
        L.ocT = L.CoutPad / 16;
        size_t n = (size_t)taps * L.ocT * L.KB * 512;
        L.wpk = (unsigned short*)alloc(n * 2);
        L.bias = (const float*)d_in[bi];
        pack_w_kernel<<<dim3((unsigned)((n + 255) / 256)), 256, 0, stream>>>(
            (const float*)d_in[wi], L.wpk, Cout, Cin, L.KB, L.ocT, taps);
        return L;
    };

    auto run = [&](const Layer& L, const unsigned short* in, int h, int w, int relu)
        -> unsigned short* {
        unsigned short* out = (unsigned short*)alloc((size_t)B * h * w * L.CoutPad * 2);
        int nPxPairs = (h * w) / 32;
        unsigned blocksX = (unsigned)((nPxPairs + 7) / 8);   // 8 waves / block
        unsigned blocksZ = (unsigned)(L.CoutPad / 32);
        int logW = 31 - __builtin_clz((unsigned)w);
        conv_wmma_kernel<<<dim3(blocksX, B, blocksZ), 256, 0, stream>>>(
            in, L.wpk, L.bias, L.Cout, out, h, w, logW, L.KB, L.ocT, L.CoutPad,
            L.taps, relu);
        return out;
    };

    auto pool = [&](const unsigned short* in, int ho, int wo, int C) -> unsigned short* {
        unsigned short* out = (unsigned short*)alloc((size_t)B * ho * wo * C * 2);
        int tot = ho * wo * C;
        pool_kernel<<<dim3((unsigned)((tot + 255) / 256), B), 256, 0, stream>>>(
            in, out, ho, wo, C);
        return out;
    };

    auto upcat = [&](const unsigned short* up, int Cu, const unsigned short* sk, int Cs,
                     int hh, int wh, int upshift) -> unsigned short* {
        int C = Cu + Cs;
        unsigned short* out = (unsigned short*)alloc((size_t)B * hh * wh * C * 2);
        int tot = hh * wh * C;
        upcat_kernel<<<dim3((unsigned)((tot + 255) / 256), B), 256, 0, stream>>>(
            up, Cu, sk, Cs, out, hh, wh, upshift);
        return out;
    };

    // --- stats + radiance mean ---
    unsigned short* stats = (unsigned short*)alloc((size_t)B * HW * 64 * 2);
    float* radmean = (float*)alloc((size_t)B * HW * 3 * 4);
    stats_kernel<<<dim3((HW + 255) / 256, B), 256, 0, stream>>>(
        color, nd, alb, spec, uvt, mv, stats, radmean, S, HW);

    // --- layers ---
    Layer pr0 = mk(bp + 0, bp + 1, 32, 36, 1);
    Layer pr1 = mk(bp + 2, bp + 3, 32, 32, 1);
    Layer pr2 = mk(bp + 4, bp + 5, 32, 32, 1);
    Layer e0  = mk(be + 0, be + 1, 64, 32, 9);
    Layer e1  = mk(be + 2, be + 3, 64, 64, 9);
    Layer e2  = mk(be + 4, be + 5, 128, 64, 9);
    Layer e3  = mk(be + 6, be + 7, 256, 128, 9);
    Layer e4  = mk(be + 8, be + 9, 512, 256, 9);
    Layer e5  = mk(be + 10, be + 11, 512, 512, 9);
    Layer bt  = mk(bb + 0, bb + 1, 512, 512, 9);
    Layer d0a = mk(bd + 0, bd + 1, 512, 1024, 9);
    Layer d0b = mk(bd + 2, bd + 3, 512, 512, 9);
    Layer d1a = mk(bd + 4, bd + 5, 256, 1024, 9);
    Layer d1b = mk(bd + 6, bd + 7, 256, 256, 9);
    Layer d2a = mk(bd + 8, bd + 9, 128, 512, 9);
    Layer d2b = mk(bd + 10, bd + 11, 128, 128, 9);
    Layer d3a = mk(bd + 12, bd + 13, 128, 256, 9);
    Layer d3b = mk(bd + 14, bd + 15, 128, 128, 9);
    Layer d4a = mk(bd + 16, bd + 17, 128, 192, 9);
    Layer d4b = mk(bd + 18, bd + 19, 128, 128, 9);
    Layer k0  = mk(bk + 0, bk + 1, 128, 160, 1);
    Layer k1  = mk(bk + 2, bk + 3, 128, 128, 1);
    Layer k2  = mk(bk + 4, bk + 5, 289, 128, 1);

    // --- pr embedding (1x1 convs) ---
    unsigned short* t1  = run(pr0, stats, H, W, 1);
    unsigned short* t2  = run(pr1, t1, H, W, 1);
    unsigned short* emb = run(pr2, t2, H, W, 1);

    // --- encoder ---
    unsigned short* a  = run(e0, emb, 256, 256, 1);
    unsigned short* s1 = run(e1, a, 256, 256, 1);
    unsigned short* p1 = pool(s1, 128, 128, 64);
    unsigned short* s2 = run(e2, p1, 128, 128, 1);
    unsigned short* p2 = pool(s2, 64, 64, 128);
    unsigned short* s3 = run(e3, p2, 64, 64, 1);
    unsigned short* p3 = pool(s3, 32, 32, 256);
    unsigned short* s4 = run(e4, p3, 32, 32, 1);
    unsigned short* p4 = pool(s4, 16, 16, 512);
    unsigned short* s5 = run(e5, p4, 16, 16, 1);
    unsigned short* p5 = pool(s5, 8, 8, 512);
    unsigned short* btm = run(bt, p5, 8, 8, 1);

    // --- decoder ---
    unsigned short* c1 = upcat(btm, 512, s5, 512, 16, 16, 1);
    unsigned short* h  = run(d0a, c1, 16, 16, 1);
    h = run(d0b, h, 16, 16, 1);
    unsigned short* c2 = upcat(h, 512, s4, 512, 32, 32, 1);
    h = run(d1a, c2, 32, 32, 1);
    h = run(d1b, h, 32, 32, 1);
    unsigned short* c3 = upcat(h, 256, s3, 256, 64, 64, 1);
    h = run(d2a, c3, 64, 64, 1);
    h = run(d2b, h, 64, 64, 1);
    unsigned short* c4 = upcat(h, 128, s2, 128, 128, 128, 1);
    h = run(d3a, c4, 128, 128, 1);
    h = run(d3b, h, 128, 128, 1);
    unsigned short* c5 = upcat(h, 128, s1, 64, 256, 256, 1);
    h = run(d4a, c5, 256, 256, 1);
    unsigned short* ctx = run(d4b, h, 256, 256, 1);

    // --- kernel-generator head ---
    unsigned short* g = upcat(emb, 32, ctx, 128, 256, 256, 0);   // plain concat
    unsigned short* q = run(k0, g, 256, 256, 1);
    q = run(k1, q, 256, 256, 1);
    unsigned short* pwb = run(k2, q, 256, 256, 0);               // no relu

    // --- softmax weights + cross-bilateral filter ---
    float* wb = (float*)alloc((size_t)B * HW * 289 * 4);
    softmax_kernel<<<dim3((HW + 255) / 256, B), 256, 0, stream>>>(pwb, wb, HW, 289, 320);
    wfilter_kernel<<<dim3((HW + 255) / 256, B), 256, 0, stream>>>(
        radmean, wb, (float*)d_out, H, W);
}
